// Qwen2VLMoTDecoderLayer_31903017075322
// MI455X (gfx1250) — compile-verified, matrix operand order checked by reading, not run on hardware
//
#include <hip/hip_runtime.h>
#include <hip/hip_bf16.h>

// ---------------- problem constants (from reference) ----------------
#define S_TOT  4096
#define HIDDEN 3584
#define NH     28
#define NKV    4
#define HD     128
#define LSEQ   1024          // S_TOT / N_SAMPLES
#define NB     4
#define KVD    (NKV*HD)      // 512
#define QD     (NH*HD)       // 3584

typedef _Float16 f16;
typedef __attribute__((ext_vector_type(16))) _Float16 v16h;
typedef __attribute__((ext_vector_type(8)))  _Float16 v8h;
typedef __attribute__((ext_vector_type(4)))  _Float16 v4h;
typedef __attribute__((ext_vector_type(8)))  float    v8f;

// ---------------- WMMA helpers ----------------
__device__ inline v8f wmma16(v16h a, v16h b, v8f c) {
  // D = A(16x32 f16) * B(32x16 f16) + C(16x16 f32)
  return __builtin_amdgcn_wmma_f32_16x16x32_f16(false, a, false, b, (short)0, c,
                                                false, false);
}

__device__ inline v8f vzero() {
  v8f z;
#pragma unroll
  for (int i = 0; i < 8; ++i) z[i] = 0.f;
  return z;
}

// 16-bit fragment load (A row / B column), per CDNA5 ISA layout:
// lane = 16*hi + r ; element K-offsets = [kbase..kbase+7] and [16+kbase..16+kbase+7]
// caller passes p already offset by (row*stride + ks + 8*hi).
__device__ inline v16h frag_ld_h(const f16* p) {
  v8h lo = *(const v8h*)p;
  v8h hi = *(const v8h*)(p + 16);
  v16h r;
#pragma unroll
  for (int i = 0; i < 8; ++i) { r[i] = lo[i]; r[8 + i] = hi[i]; }
  return r;
}

// same fragment pattern but source is f32 (exp-scores in LDS) -> convert+pack
__device__ inline v16h frag_ld_f32(const float* p) {
  v16h r;
#pragma unroll
  for (int i = 0; i < 8; ++i) { r[i] = (f16)p[i]; r[8 + i] = (f16)p[16 + i]; }
  return r;
}

// ---------------- routed GEMM:  out[rows] = X[rows] @ W^T (+bias) ----------------
// tile: 128(M) x 128(N), KSTEP=64, 8 waves -> each wave 32x64 (2x4 WMMA frags)
// Software-pipelined: global->regs for stage k+1 overlaps WMMAs on stage k.
#define GSTR 72   // LDS row stride in halves (64 + 8 pad, keeps 16B alignment)
#define GTILE (128 * GSTR)

template <typename T> struct Stage;
template <> struct Stage<float> { float4 v[8]; };
template <> struct Stage<f16>   { v8h   v[4]; };

// ---- X operand: global -> registers (with routing gather) ----
__device__ inline void ld_stage(Stage<float>& s, const float* src, int ld,
                                const int* rows, int rowbase, int nrows,
                                int kk, int tid) {
  int cid = tid & 15, r0 = tid >> 4;
#pragma unroll
  for (int p = 0; p < 8; ++p) {
    int ridx = rowbase + r0 + p * 16;
    int tok = rows ? rows[min(ridx, nrows - 1)] : min(ridx, nrows - 1);
    const float* gp = src + (size_t)tok * ld + kk + cid * 4;
    s.v[p] = *(const float4*)gp;
    if (kk + 128 < ld) __builtin_prefetch(gp + 128, 0, 0);  // global_prefetch
  }
}
__device__ inline void ld_stage(Stage<f16>& s, const f16* src, int ld,
                                const int* rows, int rowbase, int nrows,
                                int kk, int tid) {
  int cid = tid & 7, r0 = tid >> 3;
#pragma unroll
  for (int p = 0; p < 4; ++p) {
    int tok = rows[min(rowbase + r0 + p * 32, nrows - 1)];
    const f16* gp = src + (size_t)tok * ld + kk + cid * 8;
    s.v[p] = *(const v8h*)gp;
    if (kk + 128 < ld) __builtin_prefetch(gp + 128, 0, 0);
  }
}

// ---- registers -> LDS (with f32->f16 conversion where needed) ----
__device__ inline void st_stage(f16* dst, const Stage<float>& s, int tid) {
  int cid = tid & 15, r0 = tid >> 4;
#pragma unroll
  for (int p = 0; p < 8; ++p) {
    int r = r0 + p * 16;
    v4h h;
    h[0] = (f16)s.v[p].x; h[1] = (f16)s.v[p].y;
    h[2] = (f16)s.v[p].z; h[3] = (f16)s.v[p].w;
    *(v4h*)(dst + r * GSTR + cid * 4) = h;
  }
}
__device__ inline void st_stage(f16* dst, const Stage<f16>& s, int tid) {
  int cid = tid & 7, r0 = tid >> 3;
#pragma unroll
  for (int p = 0; p < 4; ++p)
    *(v8h*)(dst + (r0 + p * 32) * GSTR + cid * 8) = s.v[p];
}

// OUT_MODE 0: float out[tok*N + n]
// OUT_MODE 1: f16 vT[( (tok/L)*KVD + n )*L + tok%L]   (transposed V for PV GEMM)
template <typename T, int OUT_MODE>
__global__ __launch_bounds__(256) void gemm_routed(
    const T* __restrict__ X, const float* __restrict__ W,
    const float* __restrict__ bias, const int* __restrict__ rows, int nrows,
    int K, int N, void* __restrict__ outp) {
  __shared__ f16 Xs[2 * GTILE];
  __shared__ f16 Ws[2 * GTILE];
  const int tid = threadIdx.x;
  const int lane = tid & 31, r15 = lane & 15, hi8 = (lane >> 4) * 8;
  const int wv = tid >> 5;
  const int wm = wv >> 1, wn = wv & 1;     // 4x2 wave grid over the tile
  const int mbase = blockIdx.x * 128;
  const int nbase = blockIdx.y * 128;

  v8f acc[2][4];
#pragma unroll
  for (int i = 0; i < 2; ++i)
#pragma unroll
    for (int j = 0; j < 4; ++j) acc[i][j] = vzero();

  Stage<T>     xs;
  Stage<float> ws;
  ld_stage(xs, X, K, rows, mbase, nrows, 0, tid);
  ld_stage(ws, W, K, (const int*)nullptr, nbase, N, 0, tid);

  for (int kk = 0; kk < K; kk += 64) {
    f16* Xb = Xs + ((kk >> 6) & 1) * GTILE;
    f16* Wb = Ws + ((kk >> 6) & 1) * GTILE;
    st_stage(Xb, xs, tid);
    st_stage(Wb, ws, tid);
    __syncthreads();
    if (kk + 64 < K) {  // issue next stage's global loads before computing
      ld_stage(xs, X, K, rows, mbase, nrows, kk + 64, tid);
      ld_stage(ws, W, K, (const int*)nullptr, nbase, N, kk + 64, tid);
    }
#pragma unroll
    for (int ks = 0; ks < 64; ks += 32) {
      v16h a[2], b[4];
#pragma unroll
      for (int fi = 0; fi < 2; ++fi)
        a[fi] = frag_ld_h(Xb + (wm * 32 + fi * 16 + r15) * GSTR + ks + hi8);
#pragma unroll
      for (int fj = 0; fj < 4; ++fj)
        b[fj] = frag_ld_h(Wb + (wn * 64 + fj * 16 + r15) * GSTR + ks + hi8);
#pragma unroll
      for (int fi = 0; fi < 2; ++fi)
#pragma unroll
        for (int fj = 0; fj < 4; ++fj) acc[fi][fj] = wmma16(a[fi], b[fj], acc[fi][fj]);
    }
    // no trailing barrier: next store targets the other buffer, and the
    // single barrier at the top of the next iteration orders reuse.
  }

  // C/D layout: vgpr e, lane -> (M = e + 8*(lane>>4), N = lane&15)
#pragma unroll
  for (int fi = 0; fi < 2; ++fi)
#pragma unroll
    for (int fj = 0; fj < 4; ++fj) {
      int gcol = nbase + wn * 64 + fj * 16 + r15;
      float bv = bias ? bias[gcol] : 0.f;
#pragma unroll
      for (int e = 0; e < 8; ++e) {
        int trow = mbase + wm * 32 + fi * 16 + e + hi8;
        if (trow < nrows) {
          int tok = rows[trow];
          float val = acc[fi][fj][e] + bv;
          if (OUT_MODE == 0) {
            ((float*)outp)[(size_t)tok * N + gcol] = val;
          } else {
            int bb = tok >> 10, ll = tok & (LSEQ - 1);
            ((f16*)outp)[((size_t)(bb * KVD + gcol)) * LSEQ + ll] = (f16)val;
          }
        }
      }
    }
}

// ---------------- RMSNorm + mROPE (per modality) ----------------
__device__ inline int mrope_axis(int d) { return d < 32 ? 0 : (d < 80 ? 1 : 2); }

__global__ __launch_bounds__(256) void norm_rope(
    const float* __restrict__ qbuf, const float* __restrict__ kbuf,
    const float* __restrict__ cosb, const float* __restrict__ sinb,
    const float* __restrict__ qnw, const float* __restrict__ knw,
    const int* __restrict__ rows, int nrows,
    f16* __restrict__ qh, f16* __restrict__ kh) {
  int row = blockIdx.x;
  if (row >= nrows) return;
  int tok = rows[row];
  int lane = threadIdx.x & 31;
  int wv = threadIdx.x >> 5;
  for (int wh = wv; wh < NH + NKV; wh += 8) {
    bool isq = wh < NH;
    const float* src = isq ? qbuf + (size_t)tok * QD + wh * HD
                           : kbuf + (size_t)tok * KVD + (wh - NH) * HD;
    const float* nw = isq ? qnw : knw;
    int d0 = lane, d1 = lane + 32, d2 = lane + 64, d3 = lane + 96;
    float x0 = src[d0], x1 = src[d1], x2 = src[d2], x3 = src[d3];
    float ss = x0 * x0 + x1 * x1 + x2 * x2 + x3 * x3;
#pragma unroll
    for (int m = 16; m >= 1; m >>= 1) ss += __shfl_xor(ss, m, 32);
    float rms = rsqrtf(ss * (1.f / HD) + 1e-6f);
    float n0 = x0 * rms * nw[d0], n1 = x1 * rms * nw[d1];
    float n2 = x2 * rms * nw[d2], n3 = x3 * rms * nw[d3];
    // rotate_half within-lane: d<64 pairs with d+64 (same lane)
    float r0 = -n2, r1 = -n3, r2 = n0, r3 = n1;
    size_t tb = (size_t)tok * HD;
    float c0 = cosb[(size_t)mrope_axis(d0) * S_TOT * HD + tb + d0];
    float c1 = cosb[(size_t)mrope_axis(d1) * S_TOT * HD + tb + d1];
    float c2 = cosb[(size_t)mrope_axis(d2) * S_TOT * HD + tb + d2];
    float c3 = cosb[(size_t)mrope_axis(d3) * S_TOT * HD + tb + d3];
    float s0 = sinb[(size_t)mrope_axis(d0) * S_TOT * HD + tb + d0];
    float s1 = sinb[(size_t)mrope_axis(d1) * S_TOT * HD + tb + d1];
    float s2 = sinb[(size_t)mrope_axis(d2) * S_TOT * HD + tb + d2];
    float s3 = sinb[(size_t)mrope_axis(d3) * S_TOT * HD + tb + d3];
    f16* dst = isq ? qh + (size_t)tok * QD + wh * HD
                   : kh + (size_t)tok * KVD + (wh - NH) * HD;
    dst[d0] = (f16)(n0 * c0 + r0 * s0);
    dst[d1] = (f16)(n1 * c1 + r1 * s1);
    dst[d2] = (f16)(n2 * c2 + r2 * s2);
    dst[d3] = (f16)(n3 * c3 + r3 * s3);
  }
}

// ---------------- block-diagonal causal attention ----------------
#define QSTR 136   // Q tile LDS stride (halves), 272B keeps 16B alignment
#define SMEM_ATTN (64 * 1024 * 4 + 64 * QSTR * 2 + (256 + 64 + 64) * 4)

__global__ __launch_bounds__(256) void attn_fa(
    const f16* __restrict__ qh, const f16* __restrict__ kh,
    const f16* __restrict__ vT, f16* __restrict__ attnout) {
  extern __shared__ __align__(16) char smem[];
  float* scores = (float*)smem;                                   // 64 x 1024 f32
  f16* Qs       = (f16*)(smem + 64 * 1024 * sizeof(float));       // 64 x QSTR
  float* red    = (float*)(Qs + 64 * QSTR);                       // 256
  float* rowmax = red + 256;                                      // 64
  float* rowsum = rowmax + 64;                                    // 64

  const int b = blockIdx.z, h = blockIdx.y, qt = blockIdx.x;
  const int qbase = qt * 64;
  const int kmax = qbase + 64;           // causal: keys <= last query in tile
  const int kvh = h / (NH / NKV);        // GQA group
  const int tid = threadIdx.x;
  const int lane = tid & 31, r15 = lane & 15, hi8 = (lane >> 4) * 8;
  const int wv = tid >> 5;

  // ---- load Q tile (f16) ----
  {
    int cid = tid & 15, r0 = tid >> 4;
#pragma unroll
    for (int p = 0; p < 4; ++p) {
      int r = r0 + p * 16;
      const f16* src = qh + (size_t)(b * LSEQ + qbase + r) * QD + h * HD + cid * 8;
      *(v8h*)(Qs + r * QSTR + cid * 8) = *(const v8h*)src;
    }
  }
  __syncthreads();

  // ---- phase 1: S = Q K^T * scale, causal-masked, into LDS ----
  {
    const int mh = wv & 1;     // rows mh*32 .. +32
    const int np = wv >> 1;    // column group (stride 64)
    v16h a[2][4];
#pragma unroll
    for (int fi = 0; fi < 2; ++fi)
#pragma unroll
      for (int ks = 0; ks < 4; ++ks)
        a[fi][ks] = frag_ld_h(Qs + (mh * 32 + fi * 16 + r15) * QSTR + ks * 32 + hi8);
    const float scale = 0.08838834764831845f;  // 1/sqrt(128)
    for (int n0 = np * 16; n0 < kmax; n0 += 64) {
      v8f c0 = vzero(), c1 = vzero();
#pragma unroll
      for (int ks = 0; ks < 4; ++ks) {
        const f16* kp =
            kh + (size_t)(b * LSEQ + n0 + r15) * KVD + kvh * HD + ks * 32 + hi8;
        v16h bf = frag_ld_h(kp);
        c0 = wmma16(a[0][ks], bf, c0);
        c1 = wmma16(a[1][ks], bf, c1);
      }
      int keyg = n0 + r15;
#pragma unroll
      for (int e = 0; e < 8; ++e) {
        int rl0 = mh * 32 + e + hi8;
        int rl1 = rl0 + 16;
        float s0 = c0[e] * scale, s1 = c1[e] * scale;
        if (keyg > qbase + rl0) s0 = -1e30f;
        if (keyg > qbase + rl1) s1 = -1e30f;
        scores[rl0 * LSEQ + keyg] = s0;
        scores[rl1 * LSEQ + keyg] = s1;
      }
    }
  }
  __syncthreads();

  // ---- phase 2: softmax (exp stored in-place, row sums kept) ----
  {
    int row = tid >> 2, qtr = tid & 3;
    int seg = kmax >> 2;
    float* sp = scores + row * LSEQ;
    float m = -3.0e38f;
    for (int c = qtr * seg; c < (qtr + 1) * seg; ++c) m = fmaxf(m, sp[c]);
    red[tid] = m;
    __syncthreads();
    if (qtr == 0)
      rowmax[row] = fmaxf(fmaxf(red[tid], red[tid + 1]),
                          fmaxf(red[tid + 2], red[tid + 3]));
    __syncthreads();
    float mm = rowmax[row];
    float s = 0.f;
    for (int c = qtr * seg; c < (qtr + 1) * seg; ++c) {
      float e = __expf(sp[c] - mm);
      sp[c] = e;
      s += e;
    }
    red[tid] = s;
    __syncthreads();
    if (qtr == 0)
      rowsum[row] = red[tid] + red[tid + 1] + red[tid + 2] + red[tid + 3];
  }
  __syncthreads();

  // ---- phase 3: O = (E @ V) / rowsum ----
  {
    int nb0 = wv * 16;  // each wave owns 16 of the 128 head dims
    v8f o[4];
#pragma unroll
    for (int fi = 0; fi < 4; ++fi) o[fi] = vzero();
    for (int kk = 0; kk < kmax; kk += 32) {
      const f16* vp =
          vT + ((size_t)(b * NKV + kvh) * HD + nb0 + r15) * LSEQ + kk + hi8;
      v16h bf = frag_ld_h(vp);
#pragma unroll
      for (int fi = 0; fi < 4; ++fi) {
        v16h af = frag_ld_f32(scores + (fi * 16 + r15) * LSEQ + kk + hi8);
        o[fi] = wmma16(af, bf, o[fi]);
      }
    }
#pragma unroll
    for (int fi = 0; fi < 4; ++fi)
#pragma unroll
      for (int e = 0; e < 8; ++e) {
        int rloc = fi * 16 + e + hi8;
        int tok = b * LSEQ + qbase + rloc;
        float val = o[fi][e] / rowsum[rloc];
        attnout[(size_t)tok * QD + h * HD + nb0 + r15] = (f16)val;
      }
  }
}

// ---------------- host launcher ----------------
extern "C" void kernel_launch(void* const* d_in, const int* in_sizes, int n_in,
                              void* d_out, int out_size, void* d_ws,
                              size_t ws_size, hipStream_t stream) {
  const float* X    = (const float*)d_in[0];
  const float* cosb = (const float*)d_in[1];
  const float* sinb = (const float*)d_in[2];
  const int* rows[2] = {(const int*)d_in[3], (const int*)d_in[4]};
  int nr[2] = {in_sizes[3], in_sizes[4]};
  // weights: und = d_in[6..14], geo = d_in[15..23]
  const float *qw[2], *qb[2], *kw[2], *kb[2], *vw[2], *vb[2], *ow[2], *qn[2], *kn[2];
  for (int m = 0; m < 2; ++m) {
    int o = 6 + m * 9;
    qw[m] = (const float*)d_in[o + 0]; qb[m] = (const float*)d_in[o + 1];
    kw[m] = (const float*)d_in[o + 2]; kb[m] = (const float*)d_in[o + 3];
    vw[m] = (const float*)d_in[o + 4]; vb[m] = (const float*)d_in[o + 5];
    ow[m] = (const float*)d_in[o + 6];
    qn[m] = (const float*)d_in[o + 7]; kn[m] = (const float*)d_in[o + 8];
  }

  // workspace carve (128 MB total)
  char* ws = (char*)d_ws;
  float* qbuf = (float*)(ws + 0);            //  4096*3584 f32 = 58,720,256
  float* kbuf = (float*)(ws + 58720256);     //  4096* 512 f32 =  8,388,608
  f16*   vT   = (f16*)(ws + 67108864);       //  transposed V f16 = 4,194,304
  f16*   qh   = (f16*)(ws + 71303168);       //  rotated Q f16 = 29,360,128
  f16*   kh   = (f16*)(ws + 100663296);      //  rotated K f16 =  4,194,304
  f16*   ao   = (f16*)(ws + 104857600);      //  attn out f16  = 29,360,128

  for (int m = 0; m < 2; ++m) {
    dim3 gq((nr[m] + 127) / 128, QD / 128);
    gemm_routed<float, 0><<<gq, 256, 0, stream>>>(X, qw[m], qb[m], rows[m],
                                                  nr[m], HIDDEN, QD, qbuf);
    dim3 gk((nr[m] + 127) / 128, KVD / 128);
    gemm_routed<float, 0><<<gk, 256, 0, stream>>>(X, kw[m], kb[m], rows[m],
                                                  nr[m], HIDDEN, KVD, kbuf);
    gemm_routed<float, 1><<<gk, 256, 0, stream>>>(X, vw[m], vb[m], rows[m],
                                                  nr[m], HIDDEN, KVD, vT);
    norm_rope<<<nr[m], 256, 0, stream>>>(qbuf, kbuf, cosb, sinb, qn[m], kn[m],
                                         rows[m], nr[m], qh, kh);
  }

  dim3 ga(LSEQ / 64, NH, NB);
  attn_fa<<<ga, 256, SMEM_ATTN, stream>>>(qh, kh, vT, ao);

  for (int m = 0; m < 2; ++m) {
    dim3 go((nr[m] + 127) / 128, HIDDEN / 128);
    gemm_routed<f16, 0><<<go, 256, 0, stream>>>(ao, ow[m], nullptr, rows[m],
                                                nr[m], QD, HIDDEN, d_out);
  }
}
